// PFAFM_13125420057357
// MI455X (gfx1250) — matrix operand order repaired and checked by reading, not compiled, not run on hardware
//
#include <hip/hip_runtime.h>
#include <math.h>

// ---------------------------------------------------------------------------
// MI455X (gfx1250) implementation.
//  * Convs with OC%16==0, K%64==0 -> implicit GEMM on v_wmma_f32_16x16x32_f16:
//    each wave computes a 16(oc) x 32(px) tile, 64 K per round -> 4 WMMAs.
//  * Double-buffered LDS pipeline: TDM (tensor_load_to_lds) stages the next
//    weight tile and vectorized gathers stage the next patch tile while the
//    current buffer feeds the WMMA pipe; one barrier per round, TENSORcnt
//    drained just before the consuming round.
//  * bias/BN/ReLU and residual adds fused into the conv epilogue (workload is
//    bandwidth bound: ~400MB unique traffic, L2-resident @ 192MB/23.3TB/s).
//  * FFT work -> LDS-resident radix-2 fp32 VALU (generic stride).
// ---------------------------------------------------------------------------

typedef __attribute__((ext_vector_type(16))) _Float16 v16h;
typedef __attribute__((ext_vector_type(8)))  _Float16 v8h;
typedef __attribute__((ext_vector_type(8)))  float    v8f;
typedef __attribute__((ext_vector_type(4)))  float    v4f;
typedef __attribute__((ext_vector_type(4)))  unsigned int u32x4;
typedef __attribute__((ext_vector_type(8)))  int          i32x8;
typedef __attribute__((ext_vector_type(4)))  int          i32x4;

#if defined(__has_builtin)
#if __has_builtin(__builtin_amdgcn_tensor_load_to_lds) && \
    __has_builtin(__builtin_amdgcn_s_wait_tensorcnt)
#define HAVE_TDM 1
#endif
#endif

#define CW_WAVES 8

static __device__ __forceinline__ float2 cmul(float2 a, float2 b) {
  return make_float2(a.x * b.x - a.y * b.y, a.x * b.y + a.y * b.x);
}

// ---------------------------------------------------------------------------
// Implicit-GEMM conv via WMMA, double-buffered.
//   out[b,oc,y,x] = sum_k w[oc, koff+k] * patch[k, (b,y,x)]
// Wave tile: 16 oc x 32 px.  ACC: accumulate into out.  POST: bias+BN+ReLU.
// res: optional residual added in epilogue.
// ---------------------------------------------------------------------------
__global__ __launch_bounds__(256)
void k_conv_wmma(const float* __restrict__ in, const float* __restrict__ wgt,
                 float* __restrict__ out,
                 const float* __restrict__ bias, const float* __restrict__ bsc,
                 const float* __restrict__ bof, const float* __restrict__ res,
                 int IC, int OC, int ldA, int KH, int DIL, int ACC, int POST,
                 int H, int W)
{
  const int HW   = H * W;
  const int KK   = KH * KH;
  const int Ktot = IC * KK;

  const int octile = blockIdx.x << 4;
  const int wave   = threadIdx.x >> 5;
  const int lane   = threadIdx.x & 31;
  const int npix0  = (blockIdx.y * CW_WAVES + wave) << 5;  // 32 px, one row

  const int b      = npix0 / HW;
  const int pxbase = npix0 % HW;
  const int yy     = pxbase / W;
  const int xx0    = pxbase % W;

  __shared__ __align__(16) float    sA[2][16][64];            // weights, f32
  __shared__ __align__(16) _Float16 sB[2][CW_WAVES][32][72];  // [pix][k] pad

  v8f acc0 = {}, acc1 = {};

  const float* inb = in + (size_t)b * IC * HW;

  // stage buffer `buf` with the 64-K chunk starting at k0
  auto stage = [&](int buf, int k0) {
    // ---- A: 16 x 64 f32 weight tile ----
#if HAVE_TDM
    if (wave == 0) {
      unsigned long long ga =
          (unsigned long long)(const void*)(wgt + (size_t)octile * ldA + k0);
      u32x4 g0 = {};
      g0[0] = 1u;                                    // count=1, user mode
      g0[1] = (unsigned)(size_t)&sA[buf][0][0];      // LDS byte offset
      g0[2] = (unsigned)(ga & 0xffffffffull);
      g0[3] = (unsigned)((ga >> 32) & 0x1ffffffull) | (2u << 30);  // type=2
      i32x8 g1 = {};
      g1[0] = 2 << 16;        // data_size = 4B
      g1[1] = 64 << 16;       // tensor_dim0[15:0] = 64
      g1[2] = 16 << 16;       // tensor_dim1[15:0] = 16
      g1[3] = 64 << 16;       // tile_dim0 = 64
      g1[4] = 16;             // tile_dim1 = 16
      g1[5] = ldA;            // tensor_dim0_stride[31:0]
#if __clang_major__ >= 23
      __builtin_amdgcn_tensor_load_to_lds(g0, g1, (i32x4)0, (i32x4)0, (i32x8)0, 0);
#else
      __builtin_amdgcn_tensor_load_to_lds(g0, g1, (i32x4)0, (i32x4)0, 0);
#endif
    }
#else
    {
      int m  = threadIdx.x >> 4;
      int kq = (threadIdx.x & 15) << 2;
      const float* wp = wgt + (size_t)(octile + m) * ldA + k0 + kq;
      *(v4f*)&sA[buf][m][kq] = *(const v4f*)wp;
    }
#endif
    // ---- B: 32 px x 64 k patch tile, per wave ----
    if (KH == 1) {
      // lane owns a k-index; 32 contiguous px -> float4 row loads
#pragma unroll
      for (int h = 0; h < 2; ++h) {
        int kk = (h << 5) + lane;
        const float* sp = inb + (size_t)(k0 + kk) * HW + pxbase;
        if (k0 + 64 < Ktot) __builtin_prefetch(sp + 64 * HW, 0, 1);
#pragma unroll
        for (int p = 0; p < 32; p += 4) {
          v4f v = *(const v4f*)(sp + p);
          sB[buf][wave][p + 0][kk] = (_Float16)v[0];
          sB[buf][wave][p + 1][kk] = (_Float16)v[1];
          sB[buf][wave][p + 2][kk] = (_Float16)v[2];
          sB[buf][wave][p + 3][kk] = (_Float16)v[3];
        }
      }
    } else {
      // lane owns one pixel; gather 64 patch elements (dilated 3x3)
      const int xme = xx0 + lane;
#pragma unroll 8
      for (int t = 0; t < 64; ++t) {
        int k  = k0 + t;
        int ic = k / KK; int r = k - ic * KK;
        int iy = yy  + (r / KH - (KH >> 1)) * DIL;
        int ix = xme + (r % KH - (KH >> 1)) * DIL;
        float v = 0.f;
        if ((unsigned)iy < (unsigned)H && (unsigned)ix < (unsigned)W)
          v = inb[(size_t)ic * HW + iy * W + ix];
        sB[buf][wave][lane][t] = (_Float16)v;
      }
    }
  };

  // prologue: stage buffer 0
  stage(0, 0);

  const int mrow = lane & 15;
  const int kba  = (lane >> 4) << 3;   // A lane K-base: 0 or 8
  const int kbb  = (lane >> 4) << 3;   // B lane K-base: 0 or 8

  for (int k0 = 0; k0 < Ktot; k0 += 64) {
    const int buf = (k0 >> 6) & 1;
#if HAVE_TDM
    if (wave == 0) __builtin_amdgcn_s_wait_tensorcnt(0);
#endif
    __syncthreads();                       // staged data visible to all
    if (k0 + 64 < Ktot) stage(buf ^ 1, k0 + 64);  // overlap with compute

    // ---- 4 WMMAs: 2 K-chunks x 2 N-tiles, A fragment shared per chunk ----
#pragma unroll
    for (int kc = 0; kc < 64; kc += 32) {
      const float* ar = &sA[buf][mrow][kc + kba];
      v4f a0 = *(const v4f*)ar;
      v4f a1 = *(const v4f*)(ar + 4);
      v4f a2 = *(const v4f*)(ar + 16);
      v4f a3 = *(const v4f*)(ar + 20);
      v16h af;
#pragma unroll
      for (int i = 0; i < 4; ++i) {
        af[i]      = (_Float16)a0[i];
        af[i + 4]  = (_Float16)a1[i];
        af[i + 8]  = (_Float16)a2[i];
        af[i + 12] = (_Float16)a3[i];
      }
      {
        const _Float16* br = &sB[buf][wave][mrow][kc + kbb];
        v8h blo = *(const v8h*)br;
        v8h bhi = *(const v8h*)(br + 16);
        v16h bf;
#pragma unroll
        for (int i = 0; i < 8; ++i) { bf[i] = blo[i]; bf[i + 8] = bhi[i]; }
        acc0 = __builtin_amdgcn_wmma_f32_16x16x32_f16(
            false, af, false, bf, (short)0, acc0, false, false);
      }
      {
        const _Float16* br = &sB[buf][wave][16 + mrow][kc + kbb];
        v8h blo = *(const v8h*)br;
        v8h bhi = *(const v8h*)(br + 16);
        v16h bf;
#pragma unroll
        for (int i = 0; i < 8; ++i) { bf[i] = blo[i]; bf[i + 8] = bhi[i]; }
        acc1 = __builtin_amdgcn_wmma_f32_16x16x32_f16(
            false, af, false, bf, (short)0, acc1, false, false);
      }
    }
  }

  // ---- epilogue: fused bias/BN/ReLU + residual ----
  const float rs = rsqrtf(1.f + 1e-5f);
  const int mbase = octile + ((lane >> 4) << 3);
#pragma unroll
  for (int j = 0; j < 2; ++j) {
    const v8f* ap = j ? &acc1 : &acc0;
    const int px = pxbase + (j << 4) + (lane & 15);
#pragma unroll
    for (int r = 0; r < 8; ++r) {
      int oc = mbase + r;
      size_t idx = ((size_t)(b * OC + oc)) * HW + px;
      float v = (*ap)[r];
      if (ACC) v += out[idx];
      if (res) v += res[idx];
      if (POST) {
        v = (v + bias[oc]) * (bsc[oc] * rs) + bof[oc];
        v = fmaxf(v, 0.f);
      }
      out[idx] = v;
    }
  }
}

__global__ void k_add2(const float* __restrict__ a, const float* __restrict__ b,
                       float* __restrict__ out, int n)
{
  int i = blockIdx.x * blockDim.x + threadIdx.x;
  if (i < n) out[i] = a[i] + b[i];
}

__global__ void k_c2z(const float* __restrict__ x, float2* __restrict__ z, int n)
{
  int i = blockIdx.x * blockDim.x + threadIdx.x;
  if (i < n) z[i] = make_float2(x[i], 0.f);
}

__global__ void k_abs(const float2* __restrict__ z, float* __restrict__ r, int n)
{
  int i = blockIdx.x * blockDim.x + threadIdx.x;
  if (i < n) { float2 v = z[i]; r[i] = sqrtf(v.x * v.x + v.y * v.y); }
}

__global__ void k_cmulg(const float2* __restrict__ f, const float* __restrict__ g,
                        float2* __restrict__ o, int n)
{
  int i = blockIdx.x * blockDim.x + threadIdx.x;
  if (i < n) { float2 v = f[i]; float s = g[i]; o[i] = make_float2(v.x * s, v.y * s); }
}

// ---------------------------------------------------------------------------
// Generic in-place radix-2 FFT, one line per workgroup, data in LDS.
// sign = -1 forward, +1 inverse.  scale multiplied on store.
// ---------------------------------------------------------------------------
__global__ void k_fft(float2* __restrict__ data, int n, int logn, int stride,
                      float sign, float scale)
{
  extern __shared__ float2 sm[];
  const int line = blockIdx.x;
  const int o = line / stride;
  const int ii = line % stride;
  const size_t base = (size_t)o * n * stride + ii;
  const int t = threadIdx.x, nt = blockDim.x;

  for (int idx = t; idx < n; idx += nt) {
    int r = (int)(__brev((unsigned)idx) >> (32 - logn));
    sm[r] = data[base + (size_t)idx * stride];
  }
  __syncthreads();
  for (int s = 1; s <= logn; ++s) {
    int mh = 1 << (s - 1), m = mh << 1;
    for (int j = t; j < (n >> 1); j += nt) {
      int grp = j >> (s - 1);
      int pos = j & (mh - 1);
      int i0 = grp * m + pos, i1 = i0 + mh;
      float ang = sign * 6.283185307179586f * (float)pos / (float)m;
      float sw, cw; __sincosf(ang, &sw, &cw);
      float2 w = make_float2(cw, sw);
      float2 u = sm[i0];
      float2 v = cmul(w, sm[i1]);
      sm[i0] = make_float2(u.x + v.x, u.y + v.y);
      sm[i1] = make_float2(u.x - v.x, u.y - v.y);
    }
    __syncthreads();
  }
  for (int idx = t; idx < n; idx += nt) {
    float2 v = sm[idx];
    data[base + (size_t)idx * stride] = make_float2(v.x * scale, v.y * scale);
  }
}

// 8-point DFT along the c axis of [b,h,c,n] (stride 16384), in place.
__global__ void k_dft8(float2* __restrict__ a, float sign, float scale, int total)
{
  int idx = blockIdx.x * blockDim.x + threadIdx.x;
  if (idx >= total) return;                 // total = B*H*16384
  int bh = idx >> 14, n = idx & 16383;
  float2* p = a + ((size_t)bh << 17) + n;
  float2 z[8];
#pragma unroll
  for (int d = 0; d < 8; ++d) z[d] = p[(size_t)d << 14];
#pragma unroll
  for (int c = 0; c < 8; ++c) {
    float2 s = make_float2(0.f, 0.f);
#pragma unroll
    for (int d = 0; d < 8; ++d) {
      float ang = sign * 0.7853981633974483f * (float)((c * d) & 7);
      float sw, cw; __sincosf(ang, &sw, &cw);
      float2 tt = cmul(make_float2(cw, sw), z[d]);
      s.x += tt.x; s.y += tt.y;
    }
    p[(size_t)c << 14] = make_float2(s.x * scale, s.y * scale);
  }
}

// 1/max(||row||,1e-12) for each of the 256 channel rows (len 16384).
__global__ void k_norm(const float2* __restrict__ f, float* __restrict__ invn)
{
  int row = blockIdx.x;
  const float2* p = f + ((size_t)row << 14);
  float s = 0.f;
  for (int n = threadIdx.x; n < 16384; n += blockDim.x) {
    float2 z = p[n]; s += z.x * z.x + z.y * z.y;
  }
  __shared__ float sm[256];
  sm[threadIdx.x] = s; __syncthreads();
  for (int off = 128; off; off >>= 1) {
    if ((int)threadIdx.x < off) sm[threadIdx.x] += sm[threadIdx.x + off];
    __syncthreads();
  }
  if (!threadIdx.x) invn[row] = 1.f / fmaxf(sqrtf(sm[0]), 1e-12f);
}

// attn[bh,c,d] = temperature[h] * invn_c*invn_d * sum_n q_c[n]*q_d[n]  (complex)
__global__ void k_gram(const float2* __restrict__ f, const float* __restrict__ invn,
                       const float* __restrict__ temp, float2* __restrict__ attn)
{
  int bh = blockIdx.x;                  // 0..31  (b*8+h)
  int t  = threadIdx.x;                 // 256
  int pr = t & 63; int c = pr >> 3, d = pr & 7;
  int seg = t >> 6;                     // 4 segments of n
  const float2* qc = f + (((size_t)bh * 8 + c) << 14);
  const float2* qd = f + (((size_t)bh * 8 + d) << 14);
  float sr = 0.f, si = 0.f;
  for (int n = seg * 4096; n < (seg + 1) * 4096; ++n) {
    float2 a = qc[n], b = qd[n];
    sr += a.x * b.x - a.y * b.y;
    si += a.x * b.y + a.y * b.x;
  }
  __shared__ float smr[256], smi[256];
  smr[t] = sr; smi[t] = si; __syncthreads();
  if (t < 64) {
    float r = smr[t] + smr[t + 64] + smr[t + 128] + smr[t + 192];
    float i = smi[t] + smi[t + 64] + smi[t + 128] + smi[t + 192];
    float sc = invn[bh * 8 + c] * invn[bh * 8 + d] * temp[bh & 7];
    attn[bh * 64 + pr] = make_float2(r * sc, i * sc);
  }
}

// separate softmax over d for real and imag parts (ref's complex "softmax").
__global__ void k_softmax8(float2* __restrict__ attn)
{
  int bh = blockIdx.x; int t = threadIdx.x;   // 16 threads
  int c = t & 7; int im = t >> 3;
  float2* row = attn + bh * 64 + c * 8;
  float v[8]; float mx = -1e30f;
#pragma unroll
  for (int d = 0; d < 8; ++d) { v[d] = im ? row[d].y : row[d].x; mx = fmaxf(mx, v[d]); }
  float s = 0.f;
#pragma unroll
  for (int d = 0; d < 8; ++d) { v[d] = __expf(v[d] - mx); s += v[d]; }
  float inv = 1.f / s;
#pragma unroll
  for (int d = 0; d < 8; ++d) { if (im) row[d].y = v[d] * inv; else row[d].x = v[d] * inv; }
}

// out[bh,c,n] = sum_d attn[bh,c,d] * f[bh,d,n]
__global__ void k_apply(const float2* __restrict__ f, const float2* __restrict__ attn,
                        float2* __restrict__ out, int total)
{
  int i = blockIdx.x * blockDim.x + threadIdx.x;
  if (i >= total) return;               // total = B*H*16384
  int bh = i >> 14, n = i & 16383;
  float2 q[8];
  const float2* fp = f + ((size_t)bh << 17) + n;
#pragma unroll
  for (int d = 0; d < 8; ++d) q[d] = fp[(size_t)d << 14];
  const float2* am = attn + bh * 64;
  float2* op = out + ((size_t)bh << 17) + n;
#pragma unroll
  for (int c = 0; c < 8; ++c) {
    float2 s = make_float2(0.f, 0.f);
#pragma unroll
    for (int d = 0; d < 8; ++d) {
      float2 tt = cmul(am[c * 8 + d], q[d]);
      s.x += tt.x; s.y += tt.y;
    }
    op[(size_t)c << 14] = s;
  }
}

// 1x1 conv 64->4 on f.real + BN + ReLU  (OC too small for WMMA tile)
__global__ void k_gate_a(const float2* __restrict__ f, const float* __restrict__ wa,
                         const float* __restrict__ ba, const float* __restrict__ sa,
                         const float* __restrict__ oa, float* __restrict__ t4)
{
  int i = blockIdx.x * blockDim.x + threadIdx.x;
  if (i >= 4 * 16384) return;
  int b = i >> 14, p = i & 16383;
  float a0 = ba[0], a1 = ba[1], a2 = ba[2], a3 = ba[3];
  for (int ic = 0; ic < 64; ++ic) {
    float fr = f[(((size_t)b * 64 + ic) << 14) + p].x;
    a0 += wa[ic] * fr; a1 += wa[64 + ic] * fr;
    a2 += wa[128 + ic] * fr; a3 += wa[192 + ic] * fr;
  }
  float rs = rsqrtf(1.f + 1e-5f);
  a0 = fmaxf(a0 * sa[0] * rs + oa[0], 0.f);
  a1 = fmaxf(a1 * sa[1] * rs + oa[1], 0.f);
  a2 = fmaxf(a2 * sa[2] * rs + oa[2], 0.f);
  a3 = fmaxf(a3 * sa[3] * rs + oa[3], 0.f);
  t4[(((size_t)b * 4 + 0) << 14) + p] = a0;
  t4[(((size_t)b * 4 + 1) << 14) + p] = a1;
  t4[(((size_t)b * 4 + 2) << 14) + p] = a2;
  t4[(((size_t)b * 4 + 3) << 14) + p] = a3;
}

// 1x1 conv 4->64 + bias + sigmoid
__global__ void k_gate_b(const float* __restrict__ t4, const float* __restrict__ wb,
                         const float* __restrict__ bb, float* __restrict__ g, int n)
{
  int i = blockIdx.x * blockDim.x + threadIdx.x;
  if (i >= n) return;                   // n = 4*64*16384, layout [b,oc,p]
  int p = i & 16383; int rest = i >> 14;
  int b = rest >> 6, oc = rest & 63;
  float a = bb[oc];
#pragma unroll
  for (int ic = 0; ic < 4; ++ic)
    a += wb[oc * 4 + ic] * t4[(((size_t)b * 4 + ic) << 14) + p];
  g[i] = 1.f / (1.f + __expf(-a));
}

// global average pool over HW for [4,128,...]  (one block per (b,c))
__global__ void k_gap(const float* __restrict__ x, float* __restrict__ out)
{
  int row = blockIdx.x;
  const float* p = x + ((size_t)row << 14);
  float s = 0.f;
  for (int n = threadIdx.x; n < 16384; n += blockDim.x) s += p[n];
  __shared__ float sm[256];
  sm[threadIdx.x] = s; __syncthreads();
  for (int off = 128; off; off >>= 1) {
    if ((int)threadIdx.x < off) sm[threadIdx.x] += sm[threadIdx.x + off];
    __syncthreads();
  }
  if (!threadIdx.x) out[row] = sm[0] * (1.f / 16384.f);
}

// conv5: 1x1 128->64 on the pooled [4,128] + BN + ReLU  -> c5[4*64]
__global__ void k_c5(const float* __restrict__ gap, const float* __restrict__ w5,
                     const float* __restrict__ b5, const float* __restrict__ s5,
                     const float* __restrict__ o5, float* __restrict__ c5)
{
  int t = threadIdx.x;
  if (t >= 256) return;
  int b = t >> 6, oc = t & 63;
  float a = b5[oc];
  for (int ic = 0; ic < 128; ++ic) a += w5[oc * 128 + ic] * gap[b * 128 + ic];
  a = a * (s5[oc] * rsqrtf(1.f + 1e-5f)) + o5[oc];
  c5[t] = fmaxf(a, 0.f);
}

__global__ void k_bcast(const float* __restrict__ c5, float* __restrict__ out, int n)
{
  int i = blockIdx.x * blockDim.x + threadIdx.x;
  if (i < n) out[i] = c5[i >> 14];
}

// final 1x1 conv 32->1 + bias  -> d_out [4,1,128,128]
__global__ void k_ob(const float* __restrict__ oa, const float* __restrict__ w,
                     const float* __restrict__ b, float* __restrict__ out)
{
  int p = blockIdx.x * blockDim.x + threadIdx.x;
  if (p >= 4 * 16384) return;
  int bb = p >> 14, q = p & 16383;
  float a = b[0];
#pragma unroll 8
  for (int ic = 0; ic < 32; ++ic)
    a += w[ic] * oa[(((size_t)bb * 32 + ic) << 14) + q];
  out[p] = a;
}

// ---------------------------------------------------------------------------
extern "C" void kernel_launch(void* const* d_in, const int* in_sizes, int n_in,
                              void* d_out, int out_size, void* d_ws, size_t ws_size,
                              hipStream_t stream)
{
  (void)in_sizes; (void)n_in; (void)out_size; (void)ws_size;
  const float* X     = (const float*)d_in[0];
  const float* Wdc   = (const float*)d_in[1];
  const float* Bdc   = (const float*)d_in[2];
  const float* Sdc   = (const float*)d_in[3];
  const float* Odc   = (const float*)d_in[4];
  const float* W1    = (const float*)d_in[5];
  const float* B1    = (const float*)d_in[6];
  const float* S1    = (const float*)d_in[7];
  const float* O1    = (const float*)d_in[8];
  const float* W2    = (const float*)d_in[9];
  const float* B2    = (const float*)d_in[10];
  const float* S2    = (const float*)d_in[11];
  const float* O2    = (const float*)d_in[12];
  const float* W3    = (const float*)d_in[13];
  const float* B3    = (const float*)d_in[14];
  const float* S3    = (const float*)d_in[15];
  const float* O3    = (const float*)d_in[16];
  const float* W4    = (const float*)d_in[17];
  const float* B4    = (const float*)d_in[18];
  const float* S4    = (const float*)d_in[19];
  const float* O4    = (const float*)d_in[20];
  const float* W5    = (const float*)d_in[21];
  const float* B5    = (const float*)d_in[22];
  const float* S5    = (const float*)d_in[23];
  const float* O5    = (const float*)d_in[24];
  const float* TEMP  = (const float*)d_in[25];
  const float* Wpo   = (const float*)d_in[26];
  const float* Wwa   = (const float*)d_in[27];
  const float* Bwa   = (const float*)d_in[28];
  const float* Swa   = (const float*)d_in[29];
  const float* Owa   = (const float*)d_in[30];
  const float* Wwb   = (const float*)d_in[31];
  const float* Bwb   = (const float*)d_in[32];
  const float* Wf    = (const float*)d_in[33];
  const float* Bf    = (const float*)d_in[34];
  const float* Sf    = (const float*)d_in[35];
  const float* Of    = (const float*)d_in[36];
  const float* Woa   = (const float*)d_in[37];
  const float* Boa   = (const float*)d_in[38];
  const float* Soa   = (const float*)d_in[39];
  const float* Ooa   = (const float*)d_in[40];
  const float* Wob   = (const float*)d_in[41];
  const float* Bob   = (const float*)d_in[42];
  float* OUT = (float*)d_out;

  // workspace partition (floats)
  const size_t C64  = (size_t)4 * 64 * 16384;
  const size_t C128 = (size_t)4 * 128 * 16384;
  float* ws = (float*)d_ws;
  size_t off = 0;
  auto take = [&](size_t n) { float* p = ws + off; off += n; return p; };
  float* p_xd  = take(C128);
  float* p_c1  = take(C64);
  float* p_t   = take(C64);
  float* p_t2  = take(C64);
  float* p_F2  = take(C64);
  float* p_F3  = take(C64);
  float* p_F4  = take(C64);
  float* p_f   = take(2 * C64);     // complex fft2(c)
  float* p_a   = take(2 * C64);     // attn-out / g*f (complex)
  float* p_of  = take(C64);
  float* p_g   = take(C64);
  float* p_ol  = take(C64);
  float* p_c5b = take(C64);
  float* p_fu  = take(C64);
  float* p_oa  = take((size_t)4 * 32 * 16384);
  float* p_t4  = take((size_t)4 * 4 * 16384);
  float* p_invn = take(256);
  float* p_attn = take(32 * 64 * 2);
  float* p_gap  = take(512);
  float* p_c5   = take(256);

  (void)hipFuncSetAttribute((const void*)k_fft,
      hipFuncAttributeMaxDynamicSharedMemorySize, 131072);

  const int H = 128, W = 128;
  auto conv = [&](const float* in, const float* w, float* out_,
                  const float* bi, const float* bs, const float* bo,
                  const float* res, int IC, int OC, int ldA, int KH, int DIL,
                  int ACC, int POST) {
    dim3 g(OC / 16, 65536 / 256);
    k_conv_wmma<<<g, 256, 0, stream>>>(in, w, out_, bi, bs, bo, res,
                                       IC, OC, ldA, KH, DIL, ACC, POST, H, W);
  };
  auto fftpass = [&](float* z, int n, int logn, int stride, float sign, float scale) {
    int lines = (int)(C64 / (size_t)n);   // total complex elems / n
    k_fft<<<lines, 256, (size_t)n * sizeof(float2), stream>>>(
        (float2*)z, n, logn, stride, sign, scale);
  };
  auto ew = [&](int n) { return dim3((n + 255) / 256); };
  const float* NUL = nullptr;

  auto branch = [&](const float* cin, float* Fout) {
    int nC = (int)C64;
    // f = fft2(cin)
    k_c2z<<<ew(nC), 256, 0, stream>>>(cin, (float2*)p_f, nC);
    fftpass(p_f, 128, 7, 1,   -1.f, 1.f);
    fftpass(p_f, 128, 7, 128, -1.f, 1.f);
    // attention over (b, heads) with q=k=normalized rows of f
    k_norm<<<256, 256, 0, stream>>>((const float2*)p_f, p_invn);
    k_gram<<<32, 256, 0, stream>>>((const float2*)p_f, p_invn, TEMP, (float2*)p_attn);
    k_softmax8<<<32, 16, 0, stream>>>((float2*)p_attn);
    int tA = 4 * 8 * 16384;
    k_apply<<<ew(tA), 256, 0, stream>>>((const float2*)p_f, (const float2*)p_attn,
                                        (float2*)p_a, tA);
    // of = abs(ifft2 over (c=8, n=16384))
    k_dft8<<<ew(tA), 256, 0, stream>>>((float2*)p_a, 1.f, 0.125f, tA);
    fftpass(p_a, 16384, 14, 1, 1.f, 1.f / 16384.f);
    k_abs<<<ew(nC), 256, 0, stream>>>((const float2*)p_a, p_of, nC);
    // gating path: g = sigmoid(conv_b(bn_relu(conv_a(f.real))))
    int tG = 4 * 16384;
    k_gate_a<<<ew(tG), 256, 0, stream>>>((const float2*)p_f, Wwa, Bwa, Swa, Owa, p_t4);
    k_gate_b<<<ew(nC), 256, 0, stream>>>(p_t4, Wwb, Bwb, p_g, nC);
    // ol = abs(ifft2(g * f)) over (h,w)
    k_cmulg<<<ew(nC), 256, 0, stream>>>((const float2*)p_f, p_g, (float2*)p_a, nC);
    fftpass(p_a, 128, 7, 1,   1.f, 1.f / 128.f);
    fftpass(p_a, 128, 7, 128, 1.f, 1.f / 128.f);
    k_abs<<<ew(nC), 256, 0, stream>>>((const float2*)p_a, p_ol, nC);
    // project_out( concat[of, ol] ) as two K=64 passes; residual fused in pass2
    conv(p_of, Wpo,      Fout, NUL, NUL, NUL, NUL, 64, 64, 128, 1, 1, 0, 0);
    conv(p_ol, Wpo + 64, Fout, NUL, NUL, NUL, cin, 64, 64, 128, 1, 1, 1, 0);
  };

  // ---- trunk ----
  conv(X, Wdc, p_xd, Bdc, Sdc, Odc, NUL, 256, 128, 2304, 3, 1, 0, 1);
  conv(p_xd, W1, p_c1, B1, S1, O1, NUL, 128, 64, 128, 1, 1, 0, 1);

  conv(p_c1, W2, p_t, B2, S2, O2, NUL, 64, 64, 576, 3, 3, 0, 1);
  branch(p_t, p_F2);

  k_add2<<<ew((int)C64), 256, 0, stream>>>(p_c1, p_F2, p_t, (int)C64);
  conv(p_t, W3, p_t2, B3, S3, O3, NUL, 64, 64, 576, 3, 5, 0, 1);
  branch(p_t2, p_F3);

  k_add2<<<ew((int)C64), 256, 0, stream>>>(p_c1, p_F3, p_t, (int)C64);
  conv(p_t, W4, p_t2, B4, S4, O4, NUL, 64, 64, 576, 3, 7, 0, 1);
  branch(p_t2, p_F4);

  // conv5: GAP -> 1x1 -> broadcast
  k_gap<<<512, 256, 0, stream>>>(p_xd, p_gap);
  k_c5<<<1, 256, 0, stream>>>(p_gap, W5, B5, S5, O5, p_c5);
  k_bcast<<<ew((int)C64), 256, 0, stream>>>(p_c5, p_c5b, (int)C64);

  // fused = bn_relu(conv_f(concat[c1,F2,F3,F4,c5])): five K=64 passes,
  // bias/BN/ReLU fused into the last pass.
  conv(p_c1,  Wf,       p_fu, NUL, NUL, NUL, NUL, 64, 64, 320, 1, 1, 0, 0);
  conv(p_F2,  Wf + 64,  p_fu, NUL, NUL, NUL, NUL, 64, 64, 320, 1, 1, 1, 0);
  conv(p_F3,  Wf + 128, p_fu, NUL, NUL, NUL, NUL, 64, 64, 320, 1, 1, 1, 0);
  conv(p_F4,  Wf + 192, p_fu, NUL, NUL, NUL, NUL, 64, 64, 320, 1, 1, 1, 0);
  conv(p_c5b, Wf + 256, p_fu, Bf, Sf, Of,   NUL, 64, 64, 320, 1, 1, 1, 1);

  // out_a 3x3 64->32 + bn_relu, then out_b 1x1 32->1
  conv(p_fu, Woa, p_oa, Boa, Soa, Ooa, NUL, 64, 32, 576, 3, 1, 0, 1);
  k_ob<<<ew(4 * 16384), 256, 0, stream>>>(p_oa, Wob, Bob, OUT);
}
